// QNet_42717744726539
// MI455X (gfx1250) — compile-verified, matrix operand order checked by reading, not run on hardware
//
#include <hip/hip_runtime.h>
#include <hip/hip_bf16.h>
#include <math.h>

// ---------------- problem constants (match reference) ----------------
#define N_NODES 22528
#define N_EDGES 360448
#define FEA     60
#define FEAP    64        // FEA padded to multiple of 64 for WMMA K
#define HID     512
#define OUTC    4
#define MA      22
#define NGRAPH  1024      // N_NODES / MA
#define FLATK   12584     // MA*(HID+FEA)
#define FLATKP  12608     // padded to multiple of 64
#define SLOPE   0.01f

typedef __attribute__((ext_vector_type(16))) __bf16 v16bf;
typedef __attribute__((ext_vector_type(8)))  float  v8f;

union Frag {
    v16bf v;
    uint4 q[2];
};

// A fragment + 4 B fragments for a 16(M)x64(N) strip, one 32-wide K step.
struct FragSet {
    Frag a;
    Frag b[4];
};

__device__ __forceinline__ void ldset(FragSet& s,
                                      const unsigned short* __restrict__ arow,
                                      const unsigned short* __restrict__ brow,
                                      size_t bstride, int k) {
    s.a.q[0] = *(const uint4*)(arow + k);
    s.a.q[1] = *(const uint4*)(arow + k + 16);
#pragma unroll
    for (int i = 0; i < 4; ++i) {
        s.b[i].q[0] = *(const uint4*)(brow + (size_t)i * bstride + k);
        s.b[i].q[1] = *(const uint4*)(brow + (size_t)i * bstride + k + 16);
    }
}

__device__ __forceinline__ void mmset(v8f acc[4], const FragSet& s) {
#pragma unroll
    for (int i = 0; i < 4; ++i) {
        acc[i] = __builtin_amdgcn_wmma_f32_16x16x32_bf16(
                     false, s.a.v, false, s.b[i].v, (short)0, acc[i], false, false);
    }
}

__device__ __forceinline__ unsigned short f2bf(float f) {
    unsigned u = __float_as_uint(f);
    u += 0x7fffu + ((u >> 16) & 1u);   // round-to-nearest-even
    return (unsigned short)(u >> 16);
}

// ---------------- tiny utility kernels ----------------
__global__ void k_fill(float* __restrict__ p, long long n, float v) {
    long long i = (long long)blockIdx.x * blockDim.x + threadIdx.x;
    if (i < n) p[i] = v;
}

// deg[col[e]] += ew[e]  (deg pre-filled with 1.0 for self loops)
__global__ void k_deg(const int* __restrict__ ei, const float* __restrict__ ew,
                      float* __restrict__ deg, int E) {
    int e = blockIdx.x * blockDim.x + threadIdx.x;
    if (e < E) atomicAdd(&deg[ei[E + e]], ew[e]);
}

__global__ void k_dis(const float* __restrict__ deg, float* __restrict__ dis, int n) {
    int i = blockIdx.x * blockDim.x + threadIdx.x;
    if (i < n) {
        float d = deg[i];
        dis[i] = d > 0.f ? rsqrtf(d) : 0.f;
    }
}

__global__ void k_norm(const int* __restrict__ ei, const float* __restrict__ ew,
                       const float* __restrict__ dis, float* __restrict__ nrm, int E) {
    int e = blockIdx.x * blockDim.x + threadIdx.x;
    if (e < E) nrm[e] = dis[ei[e]] * ew[e] * dis[ei[E + e]];
}

// ---- aggregation, one wave32 per (edge|self-loop), 512-wide features ----
// out[col] += s * in[row]; coalesced float4 reads, fp32 global atomics.
__global__ __launch_bounds__(256) void k_agg_hid(
        const int* __restrict__ ei, const float* __restrict__ nrm,
        const float* __restrict__ dis,
        const float* __restrict__ in, float* __restrict__ out,
        int E, int total) {
    int w    = (int)(((long long)blockIdx.x * blockDim.x + threadIdx.x) >> 5);
    int lane = threadIdx.x & 31;
    if (w >= total) return;
    int r, c; float s;
    if (w < E) { r = ei[w]; c = ei[E + w]; s = nrm[w]; }
    else       { int i = w - E; r = i; c = i; float d = dis[i]; s = d * d; }
    const float4* pin  = (const float4*)(in + (size_t)r * HID);
    float*        pout = out + (size_t)c * HID;
#pragma unroll
    for (int it = 0; it < 4; ++it) {
        int q = lane + it * 32;
        float4 v = pin[q];
        int f = q << 2;
        atomicAdd(pout + f + 0, s * v.x);
        atomicAdd(pout + f + 1, s * v.y);
        atomicAdd(pout + f + 2, s * v.z);
        atomicAdd(pout + f + 3, s * v.w);
    }
}

// ---- aggregation for 60-wide node features (x rows are 240B = 16B aligned) ----
__global__ __launch_bounds__(256) void k_agg_fea(
        const int* __restrict__ ei, const float* __restrict__ nrm,
        const float* __restrict__ dis,
        const float* __restrict__ in, float* __restrict__ out,
        int E, int total) {
    int w    = (int)(((long long)blockIdx.x * blockDim.x + threadIdx.x) >> 5);
    int lane = threadIdx.x & 31;
    if (w >= total) return;
    int r, c; float s;
    if (w < E) { r = ei[w]; c = ei[E + w]; s = nrm[w]; }
    else       { int i = w - E; r = i; c = i; float d = dis[i]; s = d * d; }
    if (lane < FEA / 4) {                     // 15 float4 chunks cover 60 feats
        const float4* pin  = (const float4*)(in + (size_t)r * FEA);
        float*        pout = out + (size_t)c * FEAP;
        float4 v = pin[lane];
        int f = lane << 2;
        atomicAdd(pout + f + 0, s * v.x);
        atomicAdd(pout + f + 1, s * v.y);
        atomicAdd(pout + f + 2, s * v.z);
        atomicAdd(pout + f + 3, s * v.w);
    }
}

// vectorized f32 -> bf16 (n multiple of 4)
__global__ void k_cvt4(const float* __restrict__ in, unsigned short* __restrict__ out,
                       long long n4) {
    long long i = (long long)blockIdx.x * blockDim.x + threadIdx.x;
    if (i >= n4) return;
    float4 v = ((const float4*)in)[i];
    ushort4 o;
    o.x = f2bf(v.x); o.y = f2bf(v.y); o.z = f2bf(v.z); o.w = f2bf(v.w);
    ((ushort4*)out)[i] = o;
}

// Wt[n][k] (bf16, K padded to KP with zeros) = W[k][n] (f32 row-major [K][Npar])
__global__ void k_wt(const float* __restrict__ W, unsigned short* __restrict__ Wt,
                     int K, int Npar, int KP, long long total) {
    long long idx = (long long)blockIdx.x * blockDim.x + threadIdx.x;
    if (idx >= total) return;
    int n = (int)(idx / KP);
    int k = (int)(idx % KP);
    Wt[idx] = (k < K) ? f2bf(W[(size_t)k * Npar + n]) : (unsigned short)0;
}

// hcat[g][k] bf16 = concat(h2[node], x[node]) reshaped, padded to FLATKP
__global__ void k_concat(const float* __restrict__ h2, const float* __restrict__ x,
                         unsigned short* __restrict__ hcat, long long total) {
    long long idx = (long long)blockIdx.x * blockDim.x + threadIdx.x;
    if (idx >= total) return;
    int g = (int)(idx / FLATKP);
    int k = (int)(idx % FLATKP);
    float v = 0.f;
    if (k < FLATK) {
        int node = g * MA + k / (HID + FEA);
        int off  = k % (HID + FEA);
        v = (off < HID) ? h2[(size_t)node * HID + off]
                        : x[(size_t)node * FEA + (off - HID)];
    }
    hcat[idx] = f2bf(v);
}

// ---------------- WMMA bf16 GEMM:  out = lrelu(A[M×KP] @ Bt[N×KP]^T + bias) ----------------
// A row-major bf16, Bt = W^T row-major bf16. Each wave computes a 16(M)x64(N)
// strip: one A fragment reused across 4 B fragments / 4 accumulators per K-step.
// Fragment sets are software-pipelined (register double buffering) so the 10
// b128 loads of the next K-step are in flight while the current 4 WMMAs run.
// KP must be a multiple of 64 (all call sites: 64 / 512 / 12608).
// Fragment layout per CDNA5 ISA: lane L holds row/col L&15,
// K-halves at kb=(L>>4)*8 .. +7 and kb+16 .. +23.
__global__ __launch_bounds__(256) void k_gemm(
        const unsigned short* __restrict__ A, const unsigned short* __restrict__ Bt,
        const float* __restrict__ bias,
        float* __restrict__ outf, unsigned short* __restrict__ outh,
        int M, int N, int KP) {
    int wave = threadIdx.x >> 5;
    int lane = threadIdx.x & 31;
    int tn4  = N >> 6;                       // N-groups of 64
    int total = (M >> 4) * tn4;
    int t = blockIdx.x * 8 + wave;
    if (t >= total) return;                  // wave-uniform: EXEC stays all-ones

    int m0 = (t / tn4) << 4;
    int n0 = (t % tn4) << 6;
    int r  = lane & 15;
    int kb = (lane >> 4) << 3;               // 0 or 8

    const unsigned short* arow = A  + (size_t)(m0 + r) * KP + kb;
    const unsigned short* brow = Bt + (size_t)(n0 + r) * KP + kb;
    const size_t bstride = (size_t)16 * KP;  // one N-tile of Bt rows

    v8f z = {0.f, 0.f, 0.f, 0.f, 0.f, 0.f, 0.f, 0.f};
    v8f acc[4];
#pragma unroll
    for (int i = 0; i < 4; ++i) acc[i] = z;

    FragSet s0, s1;
    ldset(s0, arow, brow, bstride, 0);
    int k = 0;
    for (; k + 64 < KP; k += 64) {
        ldset(s1, arow, brow, bstride, k + 32);
        mmset(acc, s0);
        ldset(s0, arow, brow, bstride, k + 64);
        mmset(acc, s1);
    }
    ldset(s1, arow, brow, bstride, k + 32);
    mmset(acc, s0);
    mmset(acc, s1);

    // C layout: lanes 0-15 -> n=lane, M=j ; lanes 16-31 -> n=lane-16, M=8+j
    int mbase = m0 + ((lane >> 4) << 3);
#pragma unroll
    for (int i = 0; i < 4; ++i) {
        int n = n0 + (i << 4) + (lane & 15);
        float bn = bias[n];
        if (outf) {
#pragma unroll
            for (int j = 0; j < 8; ++j) {
                float v = acc[i][j] + bn;
                v = v > 0.f ? v : SLOPE * v;
                outf[(size_t)(mbase + j) * N + n] = v;
            }
        }
        if (outh) {
#pragma unroll
            for (int j = 0; j < 8; ++j) {
                float v = acc[i][j] + bn;
                v = v > 0.f ? v : SLOPE * v;
                outh[(size_t)(mbase + j) * N + n] = f2bf(v);
            }
        }
    }
}

// ---------------- output head: 512x4 linear + log_softmax, one thread per graph ----------------
__global__ void k_head(const float* __restrict__ m3, const float* __restrict__ Wo,
                       const float* __restrict__ bo, float* __restrict__ out, int G) {
    int g = blockIdx.x * blockDim.x + threadIdx.x;
    if (g >= G) return;
    float a0 = bo[0], a1 = bo[1], a2 = bo[2], a3 = bo[3];
    const float* h = m3 + (size_t)g * HID;
    for (int k = 0; k < HID; ++k) {
        float v = h[k];
        const float* w = Wo + k * OUTC;
        a0 += v * w[0]; a1 += v * w[1]; a2 += v * w[2]; a3 += v * w[3];
    }
    float mx = fmaxf(fmaxf(a0, a1), fmaxf(a2, a3));
    float s  = expf(a0 - mx) + expf(a1 - mx) + expf(a2 - mx) + expf(a3 - mx);
    float l  = logf(s) + mx;
    float* o = out + (size_t)g * OUTC;
    o[0] = a0 - l; o[1] = a1 - l; o[2] = a2 - l; o[3] = a3 - l;
}

// ---------------- launcher ----------------
static inline int nblk(long long n, int b) { return (int)((n + b - 1) / b); }

extern "C" void kernel_launch(void* const* d_in, const int* in_sizes, int n_in,
                              void* d_out, int out_size, void* d_ws, size_t ws_size,
                              hipStream_t stream) {
    (void)in_sizes; (void)n_in; (void)out_size; (void)ws_size;
    const float* x   = (const float*)d_in[0];
    const int*   ei  = (const int*)  d_in[1];   // [2][E]: row then col
    const float* ew  = (const float*)d_in[2];
    const float* W1  = (const float*)d_in[3];
    const float* b1  = (const float*)d_in[4];
    const float* W2  = (const float*)d_in[5];
    const float* b2  = (const float*)d_in[6];
    const float* Wf0 = (const float*)d_in[7];
    const float* bf0 = (const float*)d_in[8];
    const float* Wf1 = (const float*)d_in[9];
    const float* bf1 = (const float*)d_in[10];
    const float* Wf2 = (const float*)d_in[11];
    const float* bf2 = (const float*)d_in[12];
    const float* Wo  = (const float*)d_in[13];
    const float* bo  = (const float*)d_in[14];
    float* out = (float*)d_out;

    char* ws = (char*)d_ws;
    size_t off = 0;
    auto alloc = [&](size_t bytes) -> char* {
        char* p = ws + off;
        off += (bytes + 255) & ~(size_t)255;
        return p;
    };

    float*          deg   = (float*)alloc((size_t)N_NODES * 4);
    float*          dis   = (float*)alloc((size_t)N_NODES * 4);
    float*          nrm   = (float*)alloc((size_t)N_EDGES * 4);
    float*          agg1f = (float*)alloc((size_t)N_NODES * FEAP * 4);
    unsigned short* agg1h = (unsigned short*)alloc((size_t)N_NODES * FEAP * 2);
    float*          h1    = (float*)alloc((size_t)N_NODES * HID * 4);
    float*          agg2f = (float*)alloc((size_t)N_NODES * HID * 4);
    unsigned short* agg2h = (unsigned short*)alloc((size_t)N_NODES * HID * 2);
    float*          h2    = (float*)alloc((size_t)N_NODES * HID * 4);
    unsigned short* hcat  = (unsigned short*)alloc((size_t)NGRAPH * FLATKP * 2);
    unsigned short* wt1   = (unsigned short*)alloc((size_t)HID * FEAP * 2);
    unsigned short* wt2   = (unsigned short*)alloc((size_t)HID * HID * 2);
    unsigned short* wtf0  = (unsigned short*)alloc((size_t)HID * FLATKP * 2);
    unsigned short* wtf1  = (unsigned short*)alloc((size_t)HID * HID * 2);
    unsigned short* wtf2  = (unsigned short*)alloc((size_t)HID * HID * 2);
    unsigned short* m1    = (unsigned short*)alloc((size_t)NGRAPH * HID * 2);
    unsigned short* m2    = (unsigned short*)alloc((size_t)NGRAPH * HID * 2);
    float*          m3    = (float*)alloc((size_t)NGRAPH * HID * 4);

    const int B = 256;
    const int EN = N_EDGES + N_NODES;           // edges + self loops

    // gcn_norm
    k_fill<<<nblk(N_NODES, B), B, 0, stream>>>(deg, N_NODES, 1.0f);   // self-loop weight
    k_fill<<<nblk((long long)N_NODES * FEAP, B), B, 0, stream>>>(agg1f, (long long)N_NODES * FEAP, 0.f);
    k_fill<<<nblk((long long)N_NODES * HID, B), B, 0, stream>>>(agg2f, (long long)N_NODES * HID, 0.f);
    k_deg <<<nblk(N_EDGES, B), B, 0, stream>>>(ei, ew, deg, N_EDGES);
    k_dis <<<nblk(N_NODES, B), B, 0, stream>>>(deg, dis, N_NODES);
    k_norm<<<nblk(N_EDGES, B), B, 0, stream>>>(ei, ew, dis, nrm, N_EDGES);

    // weight transpose+pad to bf16 [N][KP]
    k_wt<<<nblk((long long)HID * FEAP,   B), B, 0, stream>>>(W1,  wt1,  FEA,   HID, FEAP,   (long long)HID * FEAP);
    k_wt<<<nblk((long long)HID * HID,    B), B, 0, stream>>>(W2,  wt2,  HID,   HID, HID,    (long long)HID * HID);
    k_wt<<<nblk((long long)HID * FLATKP, B), B, 0, stream>>>(Wf0, wtf0, FLATK, HID, FLATKP, (long long)HID * FLATKP);
    k_wt<<<nblk((long long)HID * HID,    B), B, 0, stream>>>(Wf1, wtf1, HID,   HID, HID,    (long long)HID * HID);
    k_wt<<<nblk((long long)HID * HID,    B), B, 0, stream>>>(Wf2, wtf2, HID,   HID, HID,    (long long)HID * HID);

    // conv1: aggregate-then-transform (A x) W1   — one wave per edge
    k_agg_fea<<<nblk((long long)EN * 32, B), B, 0, stream>>>(ei, nrm, dis, x, agg1f, N_EDGES, EN);
    k_cvt4<<<nblk((long long)N_NODES * FEAP / 4, B), B, 0, stream>>>(agg1f, agg1h, (long long)N_NODES * FEAP / 4);
    {
        int tiles = (N_NODES >> 4) * (HID >> 6);
        k_gemm<<<(tiles + 7) / 8, 256, 0, stream>>>(agg1h, wt1, b1, h1, nullptr, N_NODES, HID, FEAP);
    }

    // conv2: (A h1) W2   — one wave per edge, coalesced float4 gather + atomic scatter
    k_agg_hid<<<nblk((long long)EN * 32, B), B, 0, stream>>>(ei, nrm, dis, h1, agg2f, N_EDGES, EN);
    k_cvt4<<<nblk((long long)N_NODES * HID / 4, B), B, 0, stream>>>(agg2f, agg2h, (long long)N_NODES * HID / 4);
    {
        int tiles = (N_NODES >> 4) * (HID >> 6);
        k_gemm<<<(tiles + 7) / 8, 256, 0, stream>>>(agg2h, wt2, b2, h2, nullptr, N_NODES, HID, HID);
    }

    // concat + reshape -> MLP
    k_concat<<<nblk((long long)NGRAPH * FLATKP, B), B, 0, stream>>>(h2, x, hcat, (long long)NGRAPH * FLATKP);
    {
        int tiles = (NGRAPH >> 4) * (HID >> 6);
        k_gemm<<<(tiles + 7) / 8, 256, 0, stream>>>(hcat, wtf0, bf0, nullptr, m1, NGRAPH, HID, FLATKP);
        k_gemm<<<(tiles + 7) / 8, 256, 0, stream>>>(m1,   wtf1, bf1, nullptr, m2, NGRAPH, HID, HID);
        k_gemm<<<(tiles + 7) / 8, 256, 0, stream>>>(m2,   wtf2, bf2, m3, nullptr, NGRAPH, HID, HID);
    }

    // output head + log_softmax
    k_head<<<nblk(NGRAPH, B), B, 0, stream>>>(m3, Wo, bo, out, NGRAPH);
}
